// BLSTMModule_22273700397549
// MI455X (gfx1250) — compile-verified, matrix operand order checked by reading, not run on hardware
//
#include <hip/hip_runtime.h>
#include <cstdint>
#include <cstddef>

// ---------------- problem constants ----------------
static constexpr int kB  = 32;
static constexpr int kS  = 128;
static constexpr int kH  = 1024;
static constexpr int kHG = 4 * kH;   // 4096
static constexpr int kV  = 32000;
static constexpr int kBH = kB * kH;  // 32768

// ---------------- WMMA types ----------------
typedef __attribute__((ext_vector_type(16))) __bf16 v16bf;
typedef __attribute__((ext_vector_type(8)))  float  v8f;

union Frag {
  v16bf v;
  uint4 q[2];
  unsigned int u[8];
};

__device__ __forceinline__ unsigned short f2bf(float f) {
  unsigned int u = __float_as_uint(f);
  u += 0x7FFFu + ((u >> 16) & 1u);   // round-to-nearest-even
  return (unsigned short)(u >> 16);
}
__device__ __forceinline__ float sigmoidf_(float x) {
  return 1.0f / (1.0f + __expf(-x));
}

// ---------------- prep kernels ----------------

// out[n*K + k] = bf16(in[k*N + n])   (transpose [K,N] -> [N,K] + convert)
__global__ void k_transpose_bf16(const float* __restrict__ in,
                                 unsigned short* __restrict__ out,
                                 int K, int N) {
  long long idx = (long long)blockIdx.x * blockDim.x + threadIdx.x;
  long long total = (long long)K * N;
  if (idx >= total) return;
  int n = (int)(idx / K);
  int k = (int)(idx % K);
  out[idx] = f2bf(in[(long long)k * N + n]);
}

__global__ void k_zero_f32(float* __restrict__ p, int n) {
  int i = blockIdx.x * blockDim.x + threadIdx.x;
  if (i < n) p[i] = 0.0f;
}

__global__ void k_zero_u16(unsigned short* __restrict__ p, int n) {
  int i = blockIdx.x * blockDim.x + threadIdx.x;
  if (i < n) p[i] = 0;
}

// emb_bf[b,s,h] = bf16(embed[x[b,s], h])
__global__ void k_embed_gather(const int* __restrict__ x,
                               const float* __restrict__ embed,
                               unsigned short* __restrict__ emb_bf) {
  size_t idx = (size_t)blockIdx.x * blockDim.x + threadIdx.x;
  if (idx >= (size_t)kB * kS * kH) return;
  int hh = (int)(idx % kH);
  int bs = (int)(idx / kH);
  int tok = x[bs];
  emb_bf[idx] = f2bf(embed[(size_t)tok * kH + hh]);
}

// ---------------- generic bf16 WMMA GEMM ----------------
// C[M,N] = A[M,K] (bf16) * Bt[N,K]^T (bf16) + bias[N], C fp32.
// Block: 256 threads (8 waves), 128x128 macro tile, K-step 32.
__global__ __launch_bounds__(256)
void k_gemm_bf16(const unsigned short* __restrict__ A,
                 const unsigned short* __restrict__ Bt,
                 const float* __restrict__ bias,
                 float* __restrict__ C,
                 int M, int N, int K) {
  __shared__ __align__(16) unsigned short lsA[128 * 32];
  __shared__ __align__(16) unsigned short lsB[128 * 32];

  const int tid  = threadIdx.x;
  const int lane = tid & 31;
  const int wid  = tid >> 5;
  const int wm   = wid & 3;      // 4 wave-rows of 32 M
  const int wn   = wid >> 2;     // 2 wave-cols of 64 N
  const int nl   = lane & 15;
  const int lh   = lane >> 4;

  const int mBlk = blockIdx.y * 128;
  const int nBlk = blockIdx.x * 128;

  v8f acc[2][4];
  const v8f vzero = {0.f, 0.f, 0.f, 0.f, 0.f, 0.f, 0.f, 0.f};
#pragma unroll
  for (int mt = 0; mt < 2; ++mt)
#pragma unroll
    for (int nt = 0; nt < 4; ++nt) acc[mt][nt] = vzero;

  const int rowL  = tid >> 1;    // 0..127
  const int chunk = tid & 1;     // 0..1 -> 16 elements each

  const uint4* ga = (const uint4*)(A  + (size_t)(mBlk + rowL) * K) + 2 * chunk;
  const uint4* gb = (const uint4*)(Bt + (size_t)(nBlk + rowL) * K) + 2 * chunk;

  for (int k0 = 0; k0 < K; k0 += 32) {
    {
      uint4* la = (uint4*)(lsA + rowL * 32 + 16 * chunk);
      la[0] = ga[0]; la[1] = ga[1];
      uint4* lb = (uint4*)(lsB + rowL * 32 + 16 * chunk);
      lb[0] = gb[0]; lb[1] = gb[1];
      // prefetch next K-block (L2-resident weight stream)
      if (k0 + 32 < K) {
        __builtin_prefetch(ga + 4, 0, 3);
        __builtin_prefetch(gb + 4, 0, 3);
      }
      ga += 4;  // 32 elements = 4 uint4 per row
      gb += 4;
    }
    __syncthreads();

    Frag fa[2], fb[4];
#pragma unroll
    for (int mt = 0; mt < 2; ++mt) {
      const uint4* p = (const uint4*)(lsA + (32 * wm + 16 * mt + nl) * 32);
      fa[mt].q[0] = p[lh];
      fa[mt].q[1] = p[2 + lh];
    }
#pragma unroll
    for (int nt = 0; nt < 4; ++nt) {
      const uint4* p = (const uint4*)(lsB + (64 * wn + 16 * nt + nl) * 32);
      fb[nt].q[0] = p[lh];
      fb[nt].q[1] = p[2 + lh];
    }
#pragma unroll
    for (int mt = 0; mt < 2; ++mt)
#pragma unroll
      for (int nt = 0; nt < 4; ++nt)
        acc[mt][nt] = __builtin_amdgcn_wmma_f32_16x16x32_bf16(
            false, fa[mt].v, false, fb[nt].v, (short)0, acc[mt][nt], false, false);
    __syncthreads();
  }

#pragma unroll
  for (int nt = 0; nt < 4; ++nt) {
    int n = nBlk + 64 * wn + 16 * nt + nl;
    float bv = bias ? bias[n] : 0.0f;
#pragma unroll
    for (int mt = 0; mt < 2; ++mt) {
      int mBase = mBlk + 32 * wm + 16 * mt + 8 * lh;
#pragma unroll
      for (int r = 0; r < 8; ++r) {
        C[(size_t)(mBase + r) * N + n] = acc[mt][nt][r] + bv;
      }
    }
  }
}

// ---------------- fused LSTM step ----------------
// One launch per timestep; grid (64 n-tiles, 2 m-tiles, 2 dirs), block 128
// (4 waves = one gate each). Wave g computes the 16x16 tile
//   gates_g = xp[:,t,gH+n] + b_hh[gH+n] + h_prev @ W_hh[:, gH+n]
// then waves exchange via LDS and do the cell update. h_prev is kept
// pre-converted to bf16 (hbf), so the K-loop is pure loads + WMMA.
__global__ __launch_bounds__(128)
void k_lstm_step(const float* __restrict__ xp_f, const float* __restrict__ xp_b,
                 const unsigned short* __restrict__ WhhT_f,
                 const unsigned short* __restrict__ WhhT_b,
                 const float* __restrict__ bhh_f, const float* __restrict__ bhh_b,
                 float* __restrict__ hbuf,          // [2 dir][B*H] fp32 (final h)
                 unsigned short* __restrict__ hbf,  // [2 dir][2 parity][B*H] bf16
                 float* __restrict__ cbuf,          // [2 dir][B*H]
                 unsigned short* __restrict__ comb, // [B][S][2H] bf16
                 int step) {
  __shared__ float gates[4][16][16];

  const int lane = threadIdx.x & 31;
  const int g    = threadIdx.x >> 5;  // gate: 0=i 1=f 2=g 3=o
  const int nl   = lane & 15;
  const int lh   = lane >> 4;
  const int n0   = blockIdx.x * 16;
  const int m0   = blockIdx.y * 16;
  const int dir  = blockIdx.z;
  const int t    = dir ? (kS - 1 - step) : step;

  const float* xp = dir ? xp_b : xp_f;
  const unsigned short* WhhT = dir ? WhhT_b : WhhT_f;
  const float* bhh = dir ? bhh_b : bhh_f;

  const int p = step & 1;
  const unsigned short* h_prev = hbf + ((size_t)dir * 2 + p) * kBH;
  unsigned short* h_next       = hbf + ((size_t)dir * 2 + (1 - p)) * kBH;
  float* hout = hbuf + (size_t)dir * kBH;
  float* c    = cbuf + (size_t)dir * kBH;

  // accumulator init: xp + b_hh
  v8f acc;
  {
    int n = g * kH + n0 + nl;
    float bv = bhh[n];
#pragma unroll
    for (int r = 0; r < 8; ++r) {
      int bb = m0 + r + 8 * lh;
      acc[r] = xp[((size_t)bb * kS + t) * kHG + n] + bv;
    }
  }

  const int m    = m0 + nl;           // batch row for A
  const int wrow = g * kH + n0 + nl;  // column of W_hh (row of WhhT)
  const uint4* hp = (const uint4*)(h_prev + (size_t)m * kH);
  const uint4* wp = (const uint4*)(WhhT + (size_t)wrow * kH);
  for (int k4 = 0; k4 < kH / 8; k4 += 4) {   // 4 uint4 (32 bf16) per iter
    Frag fa, fb;
    fa.q[0] = hp[k4 + lh];
    fa.q[1] = hp[k4 + 2 + lh];
    fb.q[0] = wp[k4 + lh];
    fb.q[1] = wp[k4 + 2 + lh];
    acc = __builtin_amdgcn_wmma_f32_16x16x32_bf16(
        false, fa.v, false, fb.v, (short)0, acc, false, false);
  }

#pragma unroll
  for (int r = 0; r < 8; ++r) gates[g][r + 8 * lh][nl] = acc[r];
  __syncthreads();

  for (int e = threadIdx.x; e < 256; e += 128) {
    int ml = e >> 4, nn = e & 15;
    int bb = m0 + ml, n = n0 + nn;
    float iv = sigmoidf_(gates[0][ml][nn]);
    float fv = sigmoidf_(gates[1][ml][nn]);
    float gv = tanhf(gates[2][ml][nn]);
    float ov = sigmoidf_(gates[3][ml][nn]);
    size_t idx = (size_t)bb * kH + n;
    float cn = fv * c[idx] + iv * gv;
    c[idx] = cn;
    float hn = ov * tanhf(cn);
    unsigned short hb = f2bf(hn);
    h_next[idx] = hb;                 // bf16 h for next step's WMMA
    hout[idx] = hn;                   // fp32 h (final-state output)
    comb[((size_t)bb * kS + t) * (2 * kH) + dir * kH + n] = hb;
  }
}

// ---------------- final states -> tail of d_out ----------------
__global__ void k_final_states(const float* __restrict__ hbuf,
                               const float* __restrict__ cbuf,
                               float* __restrict__ out) {
  int i = blockIdx.x * blockDim.x + threadIdx.x;
  if (i >= 4 * kBH) return;
  int seg = i / kBH, j = i % kBH;
  float v;
  if (seg == 0)      v = hbuf[j];            // hf
  else if (seg == 1) v = cbuf[j];            // cf
  else if (seg == 2) v = hbuf[kBH + j];      // hb
  else               v = cbuf[kBH + j];      // cb
  out[(size_t)kB * kS * kV + i] = v;
}

// ---------------- host launch ----------------
extern "C" void kernel_launch(void* const* d_in, const int* in_sizes, int n_in,
                              void* d_out, int out_size, void* d_ws, size_t ws_size,
                              hipStream_t stream) {
  const int*   x      = (const int*)  d_in[0];
  const float* embed  = (const float*)d_in[1];
  const float* W_ih_f = (const float*)d_in[2];
  const float* b_ih_f = (const float*)d_in[3];
  const float* W_hh_f = (const float*)d_in[4];
  const float* b_hh_f = (const float*)d_in[5];
  const float* W_ih_b = (const float*)d_in[6];
  const float* b_ih_b = (const float*)d_in[7];
  const float* W_hh_b = (const float*)d_in[8];
  const float* b_hh_b = (const float*)d_in[9];
  const float* W_out  = (const float*)d_in[10];
  const float* b_out  = (const float*)d_in[11];
  float* out = (float*)d_out;

  char* ws = (char*)d_ws;
  size_t off = 0;
  auto take = [&](size_t bytes) -> void* {
    void* p = ws + off;
    off += (bytes + 255) & ~(size_t)255;
    return p;
  };

  unsigned short* WihT_f = (unsigned short*)take((size_t)kHG * kH * 2);
  unsigned short* WhhT_f = (unsigned short*)take((size_t)kHG * kH * 2);
  unsigned short* WihT_b = (unsigned short*)take((size_t)kHG * kH * 2);
  unsigned short* WhhT_b = (unsigned short*)take((size_t)kHG * kH * 2);
  unsigned short* WoutT  = (unsigned short*)take((size_t)kV * (2 * kH) * 2);
  unsigned short* emb_bf = (unsigned short*)take((size_t)kB * kS * kH * 2);
  float* xp_f            = (float*)take((size_t)kB * kS * kHG * 4);
  float* xp_b            = (float*)take((size_t)kB * kS * kHG * 4);
  unsigned short* comb   = (unsigned short*)take((size_t)kB * kS * (2 * kH) * 2);
  float* hbuf            = (float*)take((size_t)2 * kBH * 4);
  unsigned short* hbf    = (unsigned short*)take((size_t)4 * kBH * 2);
  float* cbuf            = (float*)take((size_t)2 * kBH * 4);

  // 1. zero recurrent state
  k_zero_f32<<<(2 * kBH + 255) / 256, 256, 0, stream>>>(hbuf, 2 * kBH);
  k_zero_u16<<<(4 * kBH + 255) / 256, 256, 0, stream>>>(hbf, 4 * kBH);
  k_zero_f32<<<(2 * kBH + 255) / 256, 256, 0, stream>>>(cbuf, 2 * kBH);

  // 2. transpose+convert weights to bf16 [N,K]
  {
    long long t1 = (long long)kH * kHG;  // 4.19M
    int g1 = (int)((t1 + 255) / 256);
    k_transpose_bf16<<<g1, 256, 0, stream>>>(W_ih_f, WihT_f, kH, kHG);
    k_transpose_bf16<<<g1, 256, 0, stream>>>(W_hh_f, WhhT_f, kH, kHG);
    k_transpose_bf16<<<g1, 256, 0, stream>>>(W_ih_b, WihT_b, kH, kHG);
    k_transpose_bf16<<<g1, 256, 0, stream>>>(W_hh_b, WhhT_b, kH, kHG);
    long long t2 = (long long)(2 * kH) * kV;  // 65.5M
    int g2 = (int)((t2 + 255) / 256);
    k_transpose_bf16<<<g2, 256, 0, stream>>>(W_out, WoutT, 2 * kH, kV);
  }

  // 3. embedding gather -> bf16
  {
    size_t tot = (size_t)kB * kS * kH;
    k_embed_gather<<<(int)((tot + 255) / 256), 256, 0, stream>>>(x, embed, emb_bf);
  }

  // 4. input projections: xp = emb @ W_ih + b_ih   (M=4096, N=4096, K=1024)
  {
    dim3 grid(kHG / 128, (kB * kS) / 128);
    k_gemm_bf16<<<grid, 256, 0, stream>>>(emb_bf, WihT_f, b_ih_f, xp_f,
                                          kB * kS, kHG, kH);
    k_gemm_bf16<<<grid, 256, 0, stream>>>(emb_bf, WihT_b, b_ih_b, xp_b,
                                          kB * kS, kHG, kH);
  }

  // 5. recurrence: 128 fused steps (both directions per launch)
  {
    dim3 grid(kH / 16, kB / 16, 2);
    for (int step = 0; step < kS; ++step) {
      k_lstm_step<<<grid, 128, 0, stream>>>(xp_f, xp_b, WhhT_f, WhhT_b,
                                            b_hh_f, b_hh_b, hbuf, hbf, cbuf,
                                            comb, step);
    }
  }

  // 6. output projection: logits = comb @ W_out + b_out (M=4096, N=32000, K=2048)
  {
    dim3 grid(kV / 128, (kB * kS) / 128);
    k_gemm_bf16<<<grid, 256, 0, stream>>>(comb, WoutT, b_out, out,
                                          kB * kS, kV, 2 * kH);
  }

  // 7. final (h, c) states appended after logits
  k_final_states<<<(4 * kBH + 255) / 256, 256, 0, stream>>>(hbuf, cbuf, out);
}
